// MultiHeadAttention_38903813767268
// MI455X (gfx1250) — matrix operand order repaired and checked by reading, not compiled, not run on hardware
//
#include <hip/hip_runtime.h>
#include <hip/hip_bf16.h>

// CDNA5 / gfx1250: wave32, WMMA 16x16x32 bf16 -> f32 accum.
typedef __attribute__((ext_vector_type(16))) __bf16 bf16x16;
typedef __attribute__((ext_vector_type(8)))  __bf16 bf16x8;
typedef __attribute__((ext_vector_type(8)))  float  f32x8;

union Frag16 { bf16x16 v; uint4 u[2]; };

__device__ __forceinline__ __bf16 f2bf(float f) { return (__bf16)f; }

__device__ __forceinline__ f32x8 wmma_bf16(bf16x16 a, bf16x16 b, f32x8 c) {
  return __builtin_amdgcn_wmma_f32_16x16x32_bf16(false, a, false, b,
                                                 (short)0, c, false, false);
}

// Async global -> LDS (CDNA5 TDM-adjacent path, ASYNCcnt-tracked).
__device__ __forceinline__ void async_b128(unsigned lds_off, const __bf16* g) {
  asm volatile("global_load_async_to_lds_b128 %0, %1, off"
               :: "v"(lds_off), "v"(g) : "memory");
}

// ---------------------------------------------------------------------------
// Kernel 0: bulk f32 -> bf16 conversion (one pass; GEMMs then read pure bf16)
// ---------------------------------------------------------------------------
__global__ __launch_bounds__(256) void cvt_kernel(
    const float* __restrict__ src, __bf16* __restrict__ dst, int n8)
{
  int i = blockIdx.x * 256 + threadIdx.x;
  if (i >= n8) return;
  const float4* s = (const float4*)src;
  float4 a = s[2 * i], b = s[2 * i + 1];
  bf16x8 r;
  r[0] = f2bf(a.x); r[1] = f2bf(a.y); r[2] = f2bf(a.z); r[3] = f2bf(a.w);
  r[4] = f2bf(b.x); r[5] = f2bf(b.y); r[6] = f2bf(b.z); r[7] = f2bf(b.w);
  *(bf16x8*)(dst + 8 * (size_t)i) = r;
}

// ---------------------------------------------------------------------------
// Kernel 1: projection  C[8192,2048] = X[8192,256] * W[2048,256]^T + bias
// X, W pre-converted bf16.  One 16x64 tile per wave (A-frag reused 4x).
// transposed==0: plain [8192,2048] (== [B,H,S',D] contiguous).
// transposed==1 (V): store as vt[B,H,D=256,S'=2048].
// ---------------------------------------------------------------------------
__global__ __launch_bounds__(256) void proj_kernel(
    const __bf16* __restrict__ X, const __bf16* __restrict__ W,
    const float* __restrict__ bias, __bf16* __restrict__ dst, int transposed)
{
  const int lane = threadIdx.x & 31;
  const int wv   = threadIdx.x >> 5;
  const int tile = blockIdx.x * 8 + wv;     // 16384 tiles = 512 x 32
  const int mt = tile >> 5;                 // 0..511
  const int nt = tile & 31;                 // 0..31 (64-wide)
  const int n  = lane & 15;
  const int hi = lane >> 4;
  const int row0 = mt * 16;
  const int col0 = nt * 64;

  const __bf16* arow = X + (size_t)(row0 + n) * 256 + hi * 8;

  f32x8 acc[4] = {};
  #pragma unroll
  for (int kk = 0; kk < 8; ++kk) {
    Frag16 a;
    a.u[0] = *(const uint4*)(arow + kk * 32);
    a.u[1] = *(const uint4*)(arow + kk * 32 + 16);
    #pragma unroll
    for (int t = 0; t < 4; ++t) {
      Frag16 bfr;
      const __bf16* brow = W + (size_t)(col0 + t * 16 + n) * 256
                             + kk * 32 + hi * 8;
      bfr.u[0] = *(const uint4*)(brow);
      bfr.u[1] = *(const uint4*)(brow + 16);
      acc[t] = wmma_bf16(a.v, bfr.v, acc[t]);
    }
  }

  #pragma unroll
  for (int t = 0; t < 4; ++t) {
    const int e = col0 + t * 16 + n;
    const float bn = bias[e];
    if (!transposed) {
      __bf16* out = dst + (size_t)e;
      #pragma unroll
      for (int v = 0; v < 8; ++v) {
        int r = row0 + v + 8 * hi;
        out[(size_t)r * 2048] = f2bf(acc[t][v] + bn);
      }
    } else {
      #pragma unroll
      for (int v = 0; v < 8; ++v) {
        int r  = row0 + v + 8 * hi;   // global row = b*2048 + s
        int bb = r >> 11;
        int s  = r & 2047;
        int h  = s >> 8;
        int sp = (s & 255) * 8 + (e >> 8);
        int d  = e & 255;
        dst[((size_t)(bb * 8 + h) * 256 + d) * 2048 + sp] = f2bf(acc[t][v] + bn);
      }
    }
  }
}

// ---------------------------------------------------------------------------
// Kernel 2: flash attention per (b,h).  Q/K: [B,H,2048,256] bf16 contiguous,
// V: vt[B,H,256,2048] bf16.  Out: ao[b][s'][h*256+d] bf16 ([B,S,2048]).
// Block = 8 waves; each wave owns 16 Q rows; 64 chunks of 32 keys, streamed
// through double-buffered LDS with global_load_async_to_lds_b128 prefetch.
// ---------------------------------------------------------------------------
// Per-thread: 8 async b128s per chunk (4 K + 4 V^T, issued in that order).
__device__ __forceinline__ void prefetch_chunk(
    const __bf16* kbase, const __bf16* vbase, int key0,
    __bf16* ksb, __bf16* vsb, int tid)
{
  const int r  = tid >> 3;
  const int cp = (tid & 7) * 32;
  const __bf16* ksrc = kbase + (size_t)(key0 + r) * 256 + cp;
  unsigned kd = (unsigned)(unsigned long long)(ksb + r * 256 + cp);
  async_b128(kd,      ksrc);
  async_b128(kd + 16, ksrc + 8);
  async_b128(kd + 32, ksrc + 16);
  async_b128(kd + 48, ksrc + 24);
  const __bf16* vsrc = vbase + (size_t)tid * 2048 + key0;
  unsigned vd = (unsigned)(unsigned long long)(vsb + tid * 32);
  async_b128(vd,      vsrc);
  async_b128(vd + 16, vsrc + 8);
  async_b128(vd + 32, vsrc + 16);
  async_b128(vd + 48, vsrc + 24);
}

__global__ __launch_bounds__(256) void attn_kernel(
    const __bf16* __restrict__ qp, const __bf16* __restrict__ kp,
    const __bf16* __restrict__ vt, __bf16* __restrict__ ao)
{
  __shared__ __bf16 ks_lds[2][32][256];   // 2 x 16 KB: K chunk (row=key)
  __shared__ __bf16 vs_lds[2][256][32];   // 2 x 16 KB: V^T chunk (row=d)
  __shared__ __bf16 p_lds[8][16][32];     //     8 KB: per-wave P staging

  const int tid  = threadIdx.x;
  const int lane = tid & 31;
  const int wv   = tid >> 5;
  const int bh   = blockIdx.x >> 4;       // 0..31
  const int qblk = blockIdx.x & 15;
  const int b = bh >> 3, h = bh & 7;
  const int n  = lane & 15;
  const int hi = lane >> 4;

  const size_t headoff = (size_t)bh * 2048 * 256;
  const __bf16* qbase = qp + headoff;
  const __bf16* kbase = kp + headoff;
  const __bf16* vbase = vt + headoff;     // [256][2048]

  const int q0 = (qblk * 8 + wv) * 16;

  // Q A-fragments (16 rows x 256 depth), resident for the whole kernel.
  Frag16 qa[8];
  const __bf16* qrow = qbase + (size_t)(q0 + n) * 256 + hi * 8;
  #pragma unroll
  for (int kk = 0; kk < 8; ++kk) {
    qa[kk].u[0] = *(const uint4*)(qrow + kk * 32);
    qa[kk].u[1] = *(const uint4*)(qrow + kk * 32 + 16);
  }

  f32x8 o[16];
  #pragma unroll
  for (int t = 0; t < 16; ++t) o[t] = (f32x8){};
  float mrow[8], lrow[8];
  #pragma unroll
  for (int v = 0; v < 8; ++v) { mrow[v] = -1e30f; lrow[v] = 0.f; }

  const float scale = 0.0625f;            // 1/sqrt(256)

  // prime the pipeline: chunk 0 -> buffer 0
  prefetch_chunk(kbase, vbase, 0, &ks_lds[0][0][0], &vs_lds[0][0][0], tid);

  for (int c = 0; c < 64; ++c) {
    const int cur = c & 1;
    if (c + 1 < 64) {
      // issue chunk c+1 into the other buffer (safe: end-of-iteration barrier
      // of c-1 guaranteed all reads of that buffer completed)
      prefetch_chunk(kbase, vbase, (c + 1) * 32,
                     &ks_lds[cur ^ 1][0][0], &vs_lds[cur ^ 1][0][0], tid);
      // drain only the oldest 8 asyncs (chunk c); keep prefetch in flight
      asm volatile("s_wait_asynccnt 8" ::: "memory");
    } else {
      asm volatile("s_wait_asynccnt 0" ::: "memory");
    }
    __syncthreads();   // all waves' chunk-c data resident in LDS

    const __bf16 (*Kc)[256] = ks_lds[cur];
    const __bf16 (*Vc)[32]  = vs_lds[cur];

    // scores: S = Q * K^T for two 16-key tiles
    f32x8 s0 = {}, s1 = {};
    #pragma unroll
    for (int kk = 0; kk < 8; ++kk) {
      Frag16 b0, b1;
      const __bf16* kr0 = &Kc[n][kk * 32 + hi * 8];
      b0.u[0] = *(const uint4*)(kr0);
      b0.u[1] = *(const uint4*)(kr0 + 16);
      const __bf16* kr1 = &Kc[16 + n][kk * 32 + hi * 8];
      b1.u[0] = *(const uint4*)(kr1);
      b1.u[1] = *(const uint4*)(kr1 + 16);
      s0 = wmma_bf16(qa[kk].v, b0.v, s0);
      s1 = wmma_bf16(qa[kk].v, b1.v, s1);
    }

    // online softmax (row stats; rows split across 16-lane halves)
    float al[8];
    #pragma unroll
    for (int v = 0; v < 8; ++v) {
      float x0 = s0[v] * scale, x1 = s1[v] * scale;
      float mx = fmaxf(x0, x1);
      #pragma unroll
      for (int off = 1; off < 16; off <<= 1)
        mx = fmaxf(mx, __shfl_xor(mx, off, 16));
      float mnew = fmaxf(mrow[v], mx);
      float a = __expf(mrow[v] - mnew);
      mrow[v] = mnew;
      float p0 = __expf(x0 - mnew);
      float p1 = __expf(x1 - mnew);
      s0[v] = p0; s1[v] = p1;
      float rs = p0 + p1;
      #pragma unroll
      for (int off = 1; off < 16; off <<= 1)
        rs += __shfl_xor(rs, off, 16);
      lrow[v] = lrow[v] * a + rs;
      al[v] = a;
    }
    #pragma unroll
    for (int t = 0; t < 16; ++t) {
      #pragma unroll
      for (int v = 0; v < 8; ++v) o[t][v] *= al[v];
    }

    // P: C-layout -> A-layout via per-wave LDS staging
    #pragma unroll
    for (int v = 0; v < 8; ++v) {
      int r = v + 8 * hi;
      p_lds[wv][r][n]      = f2bf(s0[v]);
      p_lds[wv][r][16 + n] = f2bf(s1[v]);
    }
    asm volatile("s_wait_dscnt 0" ::: "memory");  // same-wave DS ordering
    Frag16 pa;
    {
      const __bf16* pr = &p_lds[wv][n][hi * 8];
      pa.u[0] = *(const uint4*)(pr);
      pa.u[1] = *(const uint4*)(pr + 16);
    }

    // O += P * V  (16 d-tiles of 16 cols)
    #pragma unroll
    for (int t = 0; t < 16; ++t) {
      Frag16 vb;
      const __bf16* vr = &Vc[t * 16 + n][hi * 8];
      vb.u[0] = *(const uint4*)(vr);
      vb.u[1] = *(const uint4*)(vr + 16);
      o[t] = wmma_bf16(pa.v, vb.v, o[t]);
    }

    __syncthreads();   // reads of buffer `cur` done -> reusable next iteration
  }

  // epilogue: normalize and store to [b][s'][h*256 + d]
  __bf16* outp = ao + (size_t)b * 2048 * 2048 + (size_t)h * 256;
  #pragma unroll
  for (int v = 0; v < 8; ++v) {
    int sp = q0 + v + 8 * hi;
    float inv = 1.0f / lrow[v];
    __bf16* orow = outp + (size_t)sp * 2048;
    #pragma unroll
    for (int t = 0; t < 16; ++t) orow[t * 16 + n] = f2bf(o[t][v] * inv);
  }
}

// ---------------------------------------------------------------------------
// Kernel 3: out[8192,256] = A[8192,2048](bf16) * Wo[256,2048]^T(bf16) + bo
// One 16x64 tile per wave; f32 output.
// ---------------------------------------------------------------------------
__global__ __launch_bounds__(256) void outproj_kernel(
    const __bf16* __restrict__ A, const __bf16* __restrict__ Wob,
    const float* __restrict__ bo, float* __restrict__ out)
{
  const int lane = threadIdx.x & 31;
  const int wv   = threadIdx.x >> 5;
  const int tile = blockIdx.x * 8 + wv;   // 2048 tiles = 512 x 4
  const int mt = tile >> 2;               // 0..511
  const int nt = tile & 3;                // 0..3 (64-wide)
  const int n  = lane & 15;
  const int hi = lane >> 4;
  const int row0 = mt * 16, col0 = nt * 64;

  const __bf16* arow = A + (size_t)(row0 + n) * 2048 + hi * 8;

  f32x8 acc[4] = {};
  #pragma unroll 2
  for (int kk = 0; kk < 2048; kk += 32) {
    Frag16 a;
    a.u[0] = *(const uint4*)(arow + kk);
    a.u[1] = *(const uint4*)(arow + kk + 16);
    #pragma unroll
    for (int t = 0; t < 4; ++t) {
      Frag16 bfr;
      const __bf16* brow = Wob + (size_t)(col0 + t * 16 + n) * 2048
                               + kk + hi * 8;
      bfr.u[0] = *(const uint4*)(brow);
      bfr.u[1] = *(const uint4*)(brow + 16);
      acc[t] = wmma_bf16(a.v, bfr.v, acc[t]);
    }
  }

  #pragma unroll
  for (int t = 0; t < 4; ++t) {
    const int dcol = col0 + t * 16 + n;
    const float bn = bo[dcol];
    float* op = out + (size_t)dcol;
    #pragma unroll
    for (int v = 0; v < 8; ++v)
      op[(size_t)(row0 + v + 8 * hi) * 256] = acc[t][v] + bn;
  }
}

// ---------------------------------------------------------------------------
extern "C" void kernel_launch(void* const* d_in, const int* in_sizes, int n_in,
                              void* d_out, int out_size, void* d_ws, size_t ws_size,
                              hipStream_t stream) {
  (void)in_sizes; (void)n_in; (void)out_size; (void)ws_size;
  const float* query = (const float*)d_in[0];
  const float* key_  = (const float*)d_in[1];
  const float* vals  = (const float*)d_in[2];
  const float* Wq = (const float*)d_in[3];
  const float* bq = (const float*)d_in[4];
  const float* Wk = (const float*)d_in[5];
  const float* bk = (const float*)d_in[6];
  const float* Wv = (const float*)d_in[7];
  const float* bv = (const float*)d_in[8];
  const float* Wo = (const float*)d_in[9];
  const float* bo = (const float*)d_in[10];
  float* out = (float*)d_out;

  const size_t NBIG = (size_t)4 * 2048 * 2048;   // 16,777,216
  const size_t NX   = (size_t)4 * 2048 * 256;    //  2,097,152
  const size_t NW   = (size_t)2048 * 256;        //    524,288

  __bf16* qp  = (__bf16*)d_ws;       // [B,H,2048,256] bf16 (proj Q)
  __bf16* kp  = qp  + NBIG;          // proj K
  __bf16* vtp = kp  + NBIG;          // proj V transposed [B,H,256,2048]
  __bf16* aop = vtp + NBIG;          // attention output [B,S,2048]
  __bf16* xq  = aop + NBIG;          // bf16 copies of activations
  __bf16* xk  = xq  + NX;
  __bf16* xv  = xk  + NX;
  __bf16* wqb = xv  + NX;            // bf16 copies of weights
  __bf16* wkb = wqb + NW;
  __bf16* wvb = wkb + NW;
  __bf16* wob = wvb + NW;

  // one-time f32 -> bf16 conversions (bandwidth-bound pre-pass)
  cvt_kernel<<<(int)(NX / 8 / 256), 256, 0, stream>>>(query, xq, (int)(NX / 8));
  cvt_kernel<<<(int)(NX / 8 / 256), 256, 0, stream>>>(key_,  xk, (int)(NX / 8));
  cvt_kernel<<<(int)(NX / 8 / 256), 256, 0, stream>>>(vals,  xv, (int)(NX / 8));
  cvt_kernel<<<(int)(NW / 8 / 256), 256, 0, stream>>>(Wq, wqb, (int)(NW / 8));
  cvt_kernel<<<(int)(NW / 8 / 256), 256, 0, stream>>>(Wk, wkb, (int)(NW / 8));
  cvt_kernel<<<(int)(NW / 8 / 256), 256, 0, stream>>>(Wv, wvb, (int)(NW / 8));
  cvt_kernel<<<(int)(NW / 8 / 256), 256, 0, stream>>>(Wo, wob, (int)(NW / 8));

  proj_kernel<<<2048, 256, 0, stream>>>(xq, wqb, bq, qp, 0);
  proj_kernel<<<2048, 256, 0, stream>>>(xk, wkb, bk, kp, 0);
  proj_kernel<<<2048, 256, 0, stream>>>(xv, wvb, bv, vtp, 1);
  attn_kernel<<<512, 256, 0, stream>>>(qp, kp, vtp, aop);
  outproj_kernel<<<256, 256, 0, stream>>>(aop, wob, bo, out);
}
